// RoutingFreeMaskedMoE_83416854823606
// MI455X (gfx1250) — matrix operand order repaired and checked by reading, not compiled, not run on hardware
//
#include <hip/hip_runtime.h>
#include <hip/hip_bf16.h>

// Problem constants (from reference)
constexpr int Bc = 4, Tc = 2048, Dc = 2048;
constexpr int Ec = 8, Rc = 8, Fc = 1024;
constexpr int Nc = Bc * Tc;                 // 8192 tokens

typedef __attribute__((ext_vector_type(16))) __bf16 v16bf;
typedef __attribute__((ext_vector_type(8)))  __bf16 v8bf;
typedef __attribute__((ext_vector_type(8)))  float  v8f;

// ---------------------------------------------------------------------------
// WMMA helpers — A/B use contiguous k order k = 16*(L/16)+i (wave32)
// ---------------------------------------------------------------------------
__device__ __forceinline__ v8f wmma_bf16(v16bf a, v16bf b, v8f c) {
  // D = A(16x32 bf16) * B(32x16 bf16) + C(16x16 f32)
  return __builtin_amdgcn_wmma_f32_16x16x32_bf16(
      /*neg_a=*/false, a, /*neg_b=*/false, b,
      /*c_mod=*/(short)0, c, /*reuse_a=*/false, /*reuse_b=*/false);
}

// A fragment: A is row-major [M,K]; base points at A[row0][k0] (tile 16x32).
// Lane L: row = L%16; elems i = K kb + i (i = 0..15),
// kb = (L/16)*16.
__device__ __forceinline__ v16bf load_frag_A(const __bf16* base, int ld, int lane) {
  const int row = lane & 15;
  const int kb  = (lane >> 4) * 16;
  const __bf16* p = base + (size_t)row * ld + kb;
  v8bf lo = *(const v8bf*)(p);
  v8bf hi = *(const v8bf*)(p + 8);
  v16bf r;
#pragma unroll
  for (int i = 0; i < 8; ++i) { r[i] = lo[i]; r[i + 8] = hi[i]; }
  return r;
}

// B fragment: source W is row-major [Nout,K] (i.e. B transposed); base points
// at W[col0][k0] (tile K=32 x Nout=16). Lane L: col = L%16; elems i = K
// (L/16)*16 + i  -> one contiguous 16-element (32B) read per lane.
__device__ __forceinline__ v16bf load_frag_B(const __bf16* base, int ld, int lane) {
  const int col = lane & 15;
  const int kb  = (lane >> 4) * 16;
  const __bf16* p = base + (size_t)col * ld + kb;
  return *(const v16bf*)p;
}

// ---------------------------------------------------------------------------
// fp32 -> bf16 conversion (grid-stride)
// ---------------------------------------------------------------------------
__global__ void cvt_bf16_kernel(const float* __restrict__ in,
                                __bf16* __restrict__ out, size_t n) {
  size_t i = (size_t)blockIdx.x * blockDim.x + threadIdx.x;
  size_t stride = (size_t)gridDim.x * blockDim.x;
  for (; i < n; i += stride) out[i] = (__bf16)in[i];
}

// ---------------------------------------------------------------------------
// Gate: fp32 low-rank norm gate.  scores = ||x . W_A[e]^T||_2 * scale - bias.
// One block per token. Writes gate_w (masked sigmoid) and score output (-inf
// where inactive).  threshold = GATE_THRESHOLD/GATE_TEMPERATURE = 0.
// ---------------------------------------------------------------------------
__global__ __launch_bounds__(256) void gate_kernel(
    const float* __restrict__ x, const float* __restrict__ W_A,
    const float* __restrict__ gate_scale, const float* __restrict__ gate_bias,
    float* __restrict__ gate_w, float* __restrict__ score_out) {
  __shared__ float xs[Dc];
  __shared__ float dots[Ec * Rc];
  const int n = blockIdx.x;
  for (int i = threadIdx.x; i < Dc; i += blockDim.x)
    xs[i] = x[(size_t)n * Dc + i];
  __syncthreads();

  // 64 (e,r) pairs, 4 threads per pair
  const int pair = threadIdx.x >> 2;
  const int sub  = threadIdx.x & 3;
  const float* w = W_A + (size_t)pair * Dc;
  float s = 0.f;
  for (int k = sub; k < Dc; k += 4) s += xs[k] * w[k];
  s += __shfl_xor(s, 1);
  s += __shfl_xor(s, 2);
  if (sub == 0) dots[pair] = s;
  __syncthreads();

  if (threadIdx.x < Ec) {
    const int e = threadIdx.x;
    float acc = 0.f;
#pragma unroll
    for (int r = 0; r < Rc; ++r) { float d = dots[e * Rc + r]; acc += d * d; }
    float sc = sqrtf(acc) * gate_scale[e] - gate_bias[e];
    bool m = (sc >= 0.0f);
    gate_w[(size_t)n * Ec + e]  = m ? (1.f / (1.f + __expf(-sc))) : 0.f;
    score_out[(size_t)n * Ec + e] = m ? sc : -__builtin_inff();
  }
}

// ---------------------------------------------------------------------------
// H = silu(x . Wg^T) * (x . Wu^T)  for one expert.  C is [N, F], K = D.
// Block: 8 waves (256 threads) -> 64(M) x 128(N) tile; wave = 32x32 (2x2 WMMA),
// dual accumulators (gate & up).
// ---------------------------------------------------------------------------
__global__ __launch_bounds__(256) void h_kernel(
    const __bf16* __restrict__ xb, const __bf16* __restrict__ wg,
    const __bf16* __restrict__ wu, __bf16* __restrict__ H) {
  const int lane = threadIdx.x & 31;
  const int wave = threadIdx.x >> 5;
  const int row0 = blockIdx.y * 64 + (wave & 1) * 32;
  const int col0 = blockIdx.x * 128 + (wave >> 1) * 32;

  v8f accG[2][2] = {}; v8f accU[2][2] = {};
#pragma unroll 2
  for (int k = 0; k < Dc; k += 32) {
    const __bf16* ab = xb + (size_t)row0 * Dc + k;
    const __bf16* gb = wg + (size_t)col0 * Dc + k;
    const __bf16* ub = wu + (size_t)col0 * Dc + k;
    __builtin_prefetch(gb + ((size_t)(lane & 15)) * Dc + 32, 0, 1);
    __builtin_prefetch(ub + ((size_t)(lane & 15)) * Dc + 32, 0, 1);

    v16bf a0 = load_frag_A(ab, Dc, lane);
    v16bf a1 = load_frag_A(ab + (size_t)16 * Dc, Dc, lane);
    v16bf g0 = load_frag_B(gb, Dc, lane);
    v16bf g1 = load_frag_B(gb + (size_t)16 * Dc, Dc, lane);
    v16bf u0 = load_frag_B(ub, Dc, lane);
    v16bf u1 = load_frag_B(ub + (size_t)16 * Dc, Dc, lane);

    accG[0][0] = wmma_bf16(a0, g0, accG[0][0]);
    accG[0][1] = wmma_bf16(a0, g1, accG[0][1]);
    accG[1][0] = wmma_bf16(a1, g0, accG[1][0]);
    accG[1][1] = wmma_bf16(a1, g1, accG[1][1]);
    accU[0][0] = wmma_bf16(a0, u0, accU[0][0]);
    accU[0][1] = wmma_bf16(a0, u1, accU[0][1]);
    accU[1][0] = wmma_bf16(a1, u0, accU[1][0]);
    accU[1][1] = wmma_bf16(a1, u1, accU[1][1]);
  }

  // Epilogue: silu(g)*u, store bf16.  C/D layout: col = lane%16,
  // row = (lane/16)*8 + i.
  const int colA = lane & 15;
  const int rowoff = (lane >> 4) * 8;
#pragma unroll
  for (int mi = 0; mi < 2; ++mi)
#pragma unroll
    for (int ni = 0; ni < 2; ++ni)
#pragma unroll
      for (int i = 0; i < 8; ++i) {
        float g = accG[mi][ni][i];
        float u = accU[mi][ni][i];
        float h = g * (1.f / (1.f + __expf(-g))) * u;
        int r = row0 + mi * 16 + rowoff + i;
        int c = col0 + ni * 16 + colA;
        H[(size_t)r * Fc + c] = (__bf16)h;
      }
}

// ---------------------------------------------------------------------------
// out[n,d] (+)= gate_w[n,e] * (H . Wd^T)[n,d]  for one expert.
// C is [N, D], K = F.  Same tiling as h_kernel.
// ---------------------------------------------------------------------------
__global__ __launch_bounds__(256) void down_kernel(
    const __bf16* __restrict__ H, const __bf16* __restrict__ wd,
    const float* __restrict__ gate_w, float* __restrict__ out,
    int expert, int first) {
  const int lane = threadIdx.x & 31;
  const int wave = threadIdx.x >> 5;
  const int row0 = blockIdx.y * 64 + (wave & 1) * 32;
  const int col0 = blockIdx.x * 128 + (wave >> 1) * 32;

  v8f acc[2][2] = {};
#pragma unroll 2
  for (int k = 0; k < Fc; k += 32) {
    const __bf16* ab = H  + (size_t)row0 * Fc + k;
    const __bf16* bb = wd + (size_t)col0 * Fc + k;
    __builtin_prefetch(bb + ((size_t)(lane & 15)) * Fc + 32, 0, 1);

    v16bf a0 = load_frag_A(ab, Fc, lane);
    v16bf a1 = load_frag_A(ab + (size_t)16 * Fc, Fc, lane);
    v16bf b0 = load_frag_B(bb, Fc, lane);
    v16bf b1 = load_frag_B(bb + (size_t)16 * Fc, Fc, lane);

    acc[0][0] = wmma_bf16(a0, b0, acc[0][0]);
    acc[0][1] = wmma_bf16(a0, b1, acc[0][1]);
    acc[1][0] = wmma_bf16(a1, b0, acc[1][0]);
    acc[1][1] = wmma_bf16(a1, b1, acc[1][1]);
  }

  const int colA = lane & 15;
  const int rowoff = (lane >> 4) * 8;
  float gw[2][8];
#pragma unroll
  for (int mi = 0; mi < 2; ++mi)
#pragma unroll
    for (int i = 0; i < 8; ++i)
      gw[mi][i] = gate_w[(size_t)(row0 + mi * 16 + rowoff + i) * Ec + expert];

#pragma unroll
  for (int mi = 0; mi < 2; ++mi)
#pragma unroll
    for (int ni = 0; ni < 2; ++ni)
#pragma unroll
      for (int i = 0; i < 8; ++i) {
        int r = row0 + mi * 16 + rowoff + i;
        int c = col0 + ni * 16 + colA;
        float v = gw[mi][i] * acc[mi][ni][i];
        size_t idx = (size_t)r * Dc + c;
        if (first) out[idx] = v; else out[idx] += v;
      }
}

// ---------------------------------------------------------------------------
// Launch
// ---------------------------------------------------------------------------
extern "C" void kernel_launch(void* const* d_in, const int* in_sizes, int n_in,
                              void* d_out, int out_size, void* d_ws, size_t ws_size,
                              hipStream_t stream) {
  const float* x          = (const float*)d_in[0];  // [N, D]
  const float* W_A        = (const float*)d_in[1];  // [E, R, D]
  const float* gate_scale = (const float*)d_in[2];  // [E]
  const float* gate_bias  = (const float*)d_in[3];  // [E]
  const float* w_gate     = (const float*)d_in[4];  // [E, F, D]
  const float* w_up       = (const float*)d_in[5];  // [E, F, D]
  const float* w_down     = (const float*)d_in[6];  // [E, D, F]

  float* out       = (float*)d_out;                  // [N, D]
  float* score_out = out + (size_t)Nc * Dc;          // [N, E]

  // Workspace layout (bf16 copies + H buffer + gate weights) ~ 150 MB
  const size_t nX  = (size_t)Nc * Dc;       // 16.7M
  const size_t nW  = (size_t)Ec * Fc * Dc;  // 16.7M per tensor
  const size_t nH  = (size_t)Nc * Fc;       // 8.4M
  __bf16* xb   = (__bf16*)d_ws;
  __bf16* wgb  = xb  + nX;
  __bf16* wub  = wgb + nW;
  __bf16* wdb  = wub + nW;
  __bf16* Hbuf = wdb + nW;
  float*  gw   = (float*)(Hbuf + nH);       // [N, E]

  // 1) fp32 -> bf16 conversions
  cvt_bf16_kernel<<<8192, 256, 0, stream>>>(x, xb, nX);
  cvt_bf16_kernel<<<8192, 256, 0, stream>>>(w_gate, wgb, nW);
  cvt_bf16_kernel<<<8192, 256, 0, stream>>>(w_up,   wub, nW);
  cvt_bf16_kernel<<<8192, 256, 0, stream>>>(w_down, wdb, nW);

  // 2) fp32 gate scores + mask + sigmoid weights
  gate_kernel<<<Nc, 256, 0, stream>>>(x, W_A, gate_scale, gate_bias, gw, score_out);

  // 3) per-expert SwiGLU + gated down-projection (stream-serialized; Hbuf reused)
  dim3 hGrid(Fc / 128, Nc / 64);
  dim3 dGrid(Dc / 128, Nc / 64);
  for (int e = 0; e < Ec; ++e) {
    h_kernel<<<hGrid, 256, 0, stream>>>(xb, wgb + (size_t)e * Fc * Dc,
                                        wub + (size_t)e * Fc * Dc, Hbuf);
    down_kernel<<<dGrid, 256, 0, stream>>>(Hbuf, wdb + (size_t)e * Dc * Fc,
                                           gw, out, e, e == 0 ? 1 : 0);
  }
}
